// RV2BEVFrustumAttn_58102317580778
// MI455X (gfx1250) — matrix operand order repaired and checked by reading, not compiled, not run on hardware
//
#include <hip/hip_runtime.h>
#include <hip/hip_bf16.h>
#include <math.h>

// ---------------------------------------------------------------------------
// RV->BEV frustum deformable attention, MI455X (gfx1250, wave32, WMMA).
// GEMMs: f16 WMMA (v_wmma_f32_16x16x32_f16), f32 accumulate.
//  - Weights staged once per block into LDS as f16; B fragments come from
//    ds_load_b128 (no VALU write -> no WMMA->VALU hazard NOPs on the B path).
//  - A fragments double-buffered from global (f32->f16 cvt is the co-exec
//    filler between back-to-back WMMAs).
//  - Wave computes 16 x (16*NT); K fully unrolled.
// Sampling/softmax/LN are VALU kernels. All intermediates live in d_ws and
// are L2-resident (<<192MB).
// ---------------------------------------------------------------------------

typedef __attribute__((ext_vector_type(16))) _Float16 v16h;
typedef __attribute__((ext_vector_type(8)))  _Float16 v8h;
typedef __attribute__((ext_vector_type(2)))  _Float16 v2h;
typedef __attribute__((ext_vector_type(8)))  float    v8f;

constexpr int Bc    = 2;
constexpr int Dm    = 128;     // model dim
constexpr int Crv   = 64;      // output channels
constexpr int Cbev  = 256;
constexpr int Hrv   = 32, Wrv = 1024;
constexpr int Hb    = 128, Wb = 128;
constexpr int HWb   = Hb * Wb;       // 16384
constexpr int NQ    = Hrv * Wrv;     // 32768  (= 1<<15)
constexpr int HEADS = 8, PTS = 6, HD = 16;

// -ln(10000)/64 for the sinusoidal PE frequencies
#define PE_NEG_LOGF (-0.14391156831212787f)

// ---------------------------------------------------------------------------
// Register-blocked WMMA GEMM:  Out[M,NN] = epi( A @ W^T + bias )
//   A: row-major [M,K]  (TRANSA=false)  or  [K,M] with row stride ldaT (true)
//   W: row-major [NN,K] (torch Linear weight layout), staged to LDS as f16
//   EPI: 0 = none, 1 = exact GELU, 2 = add sinusoidal BEV PE (row=pixel)
//   NT: N-tiles per wave (wave computes 16 x 16*NT); K, NN compile-time.
// Fragment layouts per CDNA5 ISA 7.12.2:
//   A 16x32 f16 : lane<16 -> K pairs {0..7 | 16..23}, lane>=16 -> +8
//   B 32x16 f16 : same K order as A (K {0..7 | 16..23}, lane>=16 -> +8), col = lane&15
//   C/D 16x16 f32: VGPR r -> row (r + 8*half), col = lane&15
// ---------------------------------------------------------------------------
template<bool TRANSA, int EPI, int NT, int K, int NN>
__global__ __launch_bounds__(128) void gemm_wmma_kernel(
    const float* __restrict__ A, const float* __restrict__ W,
    const float* __restrict__ bias, float* __restrict__ Out,
    int M, int ldaT, size_t strideA, size_t strideO)
{
  constexpr int KP = K + 8;                  // padded row (halves): bank-spread
  __shared__ _Float16 wlds[NN * KP];

  // --- Cooperative W -> f16 LDS stage (once per block; W is L2-resident) ---
  for (int e = threadIdx.x * 2; e < NN * K; e += 256) {
    const int row = e / K;                    // K is a power of two
    const int col = e - row * K;
    const float2 p = *(const float2*)(W + (size_t)row * K + col);
    v2h h; h[0] = (_Float16)p.x; h[1] = (_Float16)p.y;
    *(v2h*)(wlds + row * KP + col) = h;
  }
  __syncthreads();

  A   += (size_t)blockIdx.z * strideA;
  Out += (size_t)blockIdx.z * strideO;

  const int wave = threadIdx.x >> 5;
  const int lane = threadIdx.x & 31;
  const int half = lane >> 4;
  const int idx  = lane & 15;

  constexpr int ngroups = NN / (16 * NT);
  const int group = blockIdx.x * 4 + wave;
  const int mt    = group / ngroups;
  const int ng    = group - mt * ngroups;
  if (mt >= (M >> 4)) return;                // no barriers below this point
  const int m0 = mt << 4;
  const int n0 = ng * (16 * NT);

  v8f acc[NT];
  #pragma unroll
  for (int t = 0; t < NT; ++t) {
    const float bval = bias ? bias[n0 + 16 * t + idx] : 0.f;
    #pragma unroll
    for (int r = 0; r < 8; ++r) acc[t][r] = bval;
  }

  const float* Arow = TRANSA ? (A + m0 + idx)
                             : (A + (size_t)(m0 + idx) * K);

  auto loadA = [&](int k0) {
    v16h a;
    #pragma unroll
    for (int r = 0; r < 8; ++r) {
      const int ka = k0 + ((r < 4) ? 2 * r : 16 + 2 * (r - 4)) + half * 8;
      if (TRANSA) {
        a[2*r]   = (_Float16)Arow[(size_t)ka       * ldaT];
        a[2*r+1] = (_Float16)Arow[(size_t)(ka + 1) * ldaT];
      } else {
        const float2 p = *(const float2*)(Arow + ka);
        a[2*r]   = (_Float16)p.x;
        a[2*r+1] = (_Float16)p.y;
      }
    }
    return a;
  };
  auto loadB = [&](int t, int k0) {
    const _Float16* wl = wlds + (n0 + 16 * t + idx) * KP + half * 8 + k0;
    const v8h lo = *(const v8h*)wl;          // ds_load_b128
    const v8h hi = *(const v8h*)(wl + 16);   // ds_load_b128
    v16h b;
    #pragma unroll
    for (int i = 0; i < 8; ++i) { b[i] = lo[i]; b[8 + i] = hi[i]; }
    return b;
  };

  // --- Software-pipelined K loop: fetch k+32 before the WMMA burst of k ---
  v16h a = loadA(0);
  v16h b[NT];
  #pragma unroll
  for (int t = 0; t < NT; ++t) b[t] = loadB(t, 0);

  #pragma unroll
  for (int k0 = 0; k0 < K; k0 += 32) {
    v16h a_n = a;
    v16h b_n[NT];
    if (k0 + 32 < K) {
      a_n = loadA(k0 + 32);
      #pragma unroll
      for (int t = 0; t < NT; ++t) b_n[t] = loadB(t, k0 + 32);
    }
    #pragma unroll
    for (int t = 0; t < NT; ++t) {
      acc[t] = __builtin_amdgcn_wmma_f32_16x16x32_f16(
          /*neg_a=*/false, a, /*neg_b=*/false, b[t],
          /*c_mod=*/(short)0, acc[t], /*reuse_a=*/false, /*reuse_b=*/false);
    }
    a = a_n;
    #pragma unroll
    for (int t = 0; t < NT; ++t) b[t] = (k0 + 32 < K) ? b_n[t] : b[t];
  }

  // --- Epilogue ---
  #pragma unroll
  for (int t = 0; t < NT; ++t) {
    float* op = Out + (size_t)(m0 + 8 * half) * NN + n0 + 16 * t + idx;
    #pragma unroll
    for (int r = 0; r < 8; ++r) {
      const int row = m0 + r + 8 * half;
      const int col = n0 + 16 * t + idx;
      float x = acc[t][r];
      if (EPI == 1) {                        // exact GELU
        x = 0.5f * x * (1.f + erff(x * 0.70710678118654752f));
      }
      if (EPI == 2) {                        // sinusoidal BEV PE (row = pixel)
        const int px = row & (Wb - 1);
        const int py = row >> 7;
        const float freq = __expf(PE_NEG_LOGF * (float)(col & 62));
        const float arg  = ((col < (Dm / 2)) ? (float)px : (float)py) * freq;
        x += (col & 1) ? __cosf(arg) : __sinf(arg);
      }
      op[(size_t)r * NN] = x;
    }
  }
}

// ---------------------------------------------------------------------------
// Softmax over the P=6 attention weights of each (b, query, head) group.
// ---------------------------------------------------------------------------
__global__ __launch_bounds__(256) void softmax6_kernel(float* __restrict__ aw,
                                                       int ngroups)
{
  const int g = blockIdx.x * 256 + threadIdx.x;
  if (g >= ngroups) return;
  float* p = aw + (size_t)g * PTS;
  float m = p[0];
  #pragma unroll
  for (int i = 1; i < PTS; ++i) m = fmaxf(m, p[i]);
  float e[PTS], s = 0.f;
  #pragma unroll
  for (int i = 0; i < PTS; ++i) { e[i] = __expf(p[i] - m); s += e[i]; }
  const float inv = 1.f / s;
  #pragma unroll
  for (int i = 0; i < PTS; ++i) p[i] = e[i] * inv;
}

// ---------------------------------------------------------------------------
// Bilinear deformable sampling. One thread = (b, query, head, 4 channels).
// vproj is [B, HW, 128] (channel = head*16 + c), fully L2-resident.
// ---------------------------------------------------------------------------
__device__ __forceinline__ void gather4(const float* __restrict__ vb,
                                        int xi, int yi, float w, float4& acc)
{
  if (xi >= 0 && xi < Wb && yi >= 0 && yi < Hb) {
    const float4 t = *(const float4*)(vb + (size_t)(yi * Wb + xi) * Dm);
    acc.x += w * t.x; acc.y += w * t.y; acc.z += w * t.z; acc.w += w * t.w;
  }
}

__global__ __launch_bounds__(256) void msda_sample_kernel(
    const float* __restrict__ vproj, const float* __restrict__ ref,
    const float* __restrict__ off,   const float* __restrict__ aw,
    float* __restrict__ out)
{
  const int tid  = blockIdx.x * 256 + threadIdx.x;
  const int c4   = tid & 3;
  const int head = (tid >> 2) & 7;
  const int bq   = tid >> 5;                 // b*NQ + nq
  if (bq >= Bc * NQ) return;
  const int b = bq >> 15;                    // NQ = 2^15

  const float rx = ref[(size_t)bq * 2 + 0];
  const float ry = ref[(size_t)bq * 2 + 1];
  const float* offp = off + (size_t)bq * (HEADS * PTS * 2) + head * (PTS * 2);
  const float* awp  = aw  + (size_t)bq * (HEADS * PTS)     + head * PTS;
  const float* vb   = vproj + ((size_t)b * HWb) * Dm + head * HD + c4 * 4;

  float4 acc = make_float4(0.f, 0.f, 0.f, 0.f);
  #pragma unroll
  for (int p = 0; p < PTS; ++p) {
    // grid_sample(align_corners=False): x = (ref + off/W)*W - 0.5
    const float x = rx * (float)Wb + offp[2 * p + 0] - 0.5f;
    const float y = ry * (float)Hb + offp[2 * p + 1] - 0.5f;
    const float x0f = floorf(x), y0f = floorf(y);
    const float fx = x - x0f, fy = y - y0f;
    const int x0 = (int)x0f, y0 = (int)y0f;
    const float a = awp[p];
    gather4(vb, x0,     y0,     a * (1.f - fx) * (1.f - fy), acc);
    gather4(vb, x0 + 1, y0,     a * fx * (1.f - fy),         acc);
    gather4(vb, x0,     y0 + 1, a * (1.f - fx) * fy,         acc);
    gather4(vb, x0 + 1, y0 + 1, a * fx * fy,                 acc);
  }
  *(float4*)(out + (size_t)bq * Dm + head * HD + c4 * 4) = acc;
}

// ---------------------------------------------------------------------------
// Residual + LayerNorm over d=128. One wave32 per row: 4 floats/lane,
// shuffle-tree reduction (wave32 native on gfx1250).
// ---------------------------------------------------------------------------
__device__ __forceinline__ float wave_sum32(float v)
{
  #pragma unroll
  for (int m = 16; m > 0; m >>= 1) v += __shfl_xor(v, m, 32);
  return v;
}

__global__ __launch_bounds__(256) void ln_add_kernel(
    const float* __restrict__ xa, const float* __restrict__ xb,
    const float* __restrict__ g,  const float* __restrict__ be,
    float* __restrict__ out, int M)
{
  const int row  = (blockIdx.x * 256 + threadIdx.x) >> 5;
  const int lane = threadIdx.x & 31;
  if (row >= M) return;

  const float4 a = *(const float4*)(xa + (size_t)row * Dm + lane * 4);
  const float4 b = *(const float4*)(xb + (size_t)row * Dm + lane * 4);
  float4 v = make_float4(a.x + b.x, a.y + b.y, a.z + b.z, a.w + b.w);

  const float mean = wave_sum32(v.x + v.y + v.z + v.w) * (1.f / (float)Dm);
  const float4 d = make_float4(v.x - mean, v.y - mean, v.z - mean, v.w - mean);
  const float var = wave_sum32(d.x * d.x + d.y * d.y + d.z * d.z + d.w * d.w)
                    * (1.f / (float)Dm);
  const float rstd = rsqrtf(var + 1e-5f);

  const float4 gg = *(const float4*)(g  + lane * 4);
  const float4 bb = *(const float4*)(be + lane * 4);
  float4 o = make_float4(d.x * rstd * gg.x + bb.x, d.y * rstd * gg.y + bb.y,
                         d.z * rstd * gg.z + bb.z, d.w * rstd * gg.w + bb.w);
  *(float4*)(out + (size_t)row * Dm + lane * 4) = o;
}

// ---------------------------------------------------------------------------
extern "C" void kernel_launch(void* const* d_in, const int* in_sizes, int n_in,
                              void* d_out, int out_size, void* d_ws, size_t ws_size,
                              hipStream_t stream)
{
  (void)in_sizes; (void)n_in; (void)out_size; (void)ws_size;
  const float* rv   = (const float*)d_in[0];
  const float* bev  = (const float*)d_in[1];
  const float* ref  = (const float*)d_in[2];
  const float* Wq   = (const float*)d_in[3];
  const float* bq   = (const float*)d_in[4];
  const float* Wv   = (const float*)d_in[5];
  const float* bvv  = (const float*)d_in[6];
  const float* soW  = (const float*)d_in[7];
  const float* sob  = (const float*)d_in[8];
  const float* awW  = (const float*)d_in[9];
  const float* awbv = (const float*)d_in[10];
  const float* vpW  = (const float*)d_in[11];
  const float* vpb  = (const float*)d_in[12];
  const float* opW  = (const float*)d_in[13];
  const float* opb  = (const float*)d_in[14];
  const float* ln1g = (const float*)d_in[15];
  const float* ln1b = (const float*)d_in[16];
  const float* f1W  = (const float*)d_in[17];
  const float* f1b  = (const float*)d_in[18];
  const float* f2W  = (const float*)d_in[19];
  const float* f2b  = (const float*)d_in[20];
  const float* ln2g = (const float*)d_in[21];
  const float* ln2b = (const float*)d_in[22];
  const float* Wo   = (const float*)d_in[23];
  const float* bo   = (const float*)d_in[24];
  float* out = (float*)d_out;
  float* ws  = (float*)d_ws;

  // Workspace layout (floats); buffers reused once dead:
  float* q     = ws;               // 8388608  : q, later y2 (LN2 out)
  float* value = ws + 8388608;     // 4194304  : value; [value..offb) reused as FFN hidden
  float* vproj = ws + 12582912;    // 4194304
  float* offb  = ws + 16777216;    // 6291456
  float* awbuf = ws + 23068672;    // 3145728
  float* sam   = ws + 26214400;    // 8388608  : sampled, later y (LN1 out)
  float* ym    = ws + 34603008;    // 8388608  : y_msda, later f2 out

  // 1) q = (rv^T) @ Wq^T + bq        per batch: M=NQ, K=128, wave = 16x64
  gemm_wmma_kernel<true, 0, 4, 128, 128><<<dim3(1024, 1, 2), 128, 0, stream>>>(
      rv, Wq, bq, q, NQ, NQ, (size_t)Dm * NQ, (size_t)NQ * Dm);

  // 2) value = (bev^T) @ Wv^T + bv + PE   per batch: M=HW, K=256 (PE fused)
  gemm_wmma_kernel<true, 2, 4, 256, 128><<<dim3(512, 1, 2), 128, 0, stream>>>(
      bev, Wv, bvv, value, HWb, HWb, (size_t)Cbev * HWb, (size_t)HWb * Dm);

  // 3) vproj = value @ vpW^T + vpb        M = B*HW
  gemm_wmma_kernel<false, 0, 4, 128, 128><<<dim3(1024, 1, 1), 128, 0, stream>>>(
      value, vpW, vpb, vproj, Bc * HWb, 0, 0, 0);

  // 4) sampling offsets: off = q @ soW^T + sob   N=96 (wave = 16x48)
  gemm_wmma_kernel<false, 0, 3, 128, 96><<<dim3(2048, 1, 1), 128, 0, stream>>>(
      q, soW, sob, offb, Bc * NQ, 0, 0, 0);

  // 5) attention logits: aw = q @ awW^T + awb    N=48 (wave = 16x48)
  gemm_wmma_kernel<false, 0, 3, 128, 48><<<dim3(1024, 1, 1), 128, 0, stream>>>(
      q, awW, awbv, awbuf, Bc * NQ, 0, 0, 0);

  // 6) softmax over P per (b,nq,head)
  softmax6_kernel<<<2048, 256, 0, stream>>>(awbuf, Bc * NQ * HEADS);

  // 7) bilinear deformable sampling -> sam [B,Nq,128]
  msda_sample_kernel<<<8192, 256, 0, stream>>>(vproj, ref, offb, awbuf, sam);

  // 8) y_msda = sam @ opW^T + opb
  gemm_wmma_kernel<false, 0, 4, 128, 128><<<dim3(2048, 1, 1), 128, 0, stream>>>(
      sam, opW, opb, ym, Bc * NQ, 0, 0, 0);

  // 9) y = LN(y_msda + q)  -> sam slot
  ln_add_kernel<<<8192, 256, 0, stream>>>(ym, q, ln1g, ln1b, sam, Bc * NQ);

  // 10) hidden = gelu(y @ f1^T + f1b) -> value slot (spans value+vproj)
  gemm_wmma_kernel<false, 1, 4, 128, 128><<<dim3(2048, 1, 1), 128, 0, stream>>>(
      sam, f1W, f1b, value, Bc * NQ, 0, 0, 0);

  // 11) f2out = hidden @ f2^T + f2b -> ym slot
  gemm_wmma_kernel<false, 0, 4, 128, 128><<<dim3(2048, 1, 1), 128, 0, stream>>>(
      value, f2W, f2b, ym, Bc * NQ, 0, 0, 0);

  // 12) y2 = LN(f2out + y) -> q slot
  ln_add_kernel<<<8192, 256, 0, stream>>>(ym, sam, ln2g, ln2b, q, Bc * NQ);

  // 13) out = y2 @ Wo^T + bo   N=64 -> d_out
  gemm_wmma_kernel<false, 0, 4, 128, 64><<<dim3(1024, 1, 1), 128, 0, stream>>>(
      q, Wo, bo, out, Bc * NQ, Crv, 0, 0);
}
